// VectorQuantizer_55276229100052
// MI455X (gfx1250) — matrix operand order repaired and checked by reading, not compiled, not run on hardware
//
#include <hip/hip_runtime.h>

typedef __attribute__((ext_vector_type(16))) _Float16 v16h;
typedef __attribute__((ext_vector_type(8)))  float    v8f;

#define NUM_EMB 1024
#define DIM     256
#define HW      1024
#define QELEMS  16777216
#define CHUNK_CODES 128          // codes staged in LDS per iteration (64 KB)

// ---------------------------------------------------------------------------
// Kernel 1: pack codebook to f16 (for WMMA B), compute |e_k|^2 in fp32,
// zero the global loss accumulator. One wave (32 lanes) per code row.
// ---------------------------------------------------------------------------
__global__ __launch_bounds__(256) void vq_prep(const float* __restrict__ emb,
                                               float* __restrict__ e2,
                                               _Float16* __restrict__ ef,
                                               float* __restrict__ acc) {
    int k    = blockIdx.x * 8 + (threadIdx.x >> 5);
    int lane = threadIdx.x & 31;
    const float* row  = emb + (size_t)k * DIM;
    _Float16*    frow = ef  + (size_t)k * DIM;
    float s = 0.f;
#pragma unroll
    for (int i = 0; i < 8; ++i) {
        int d = i * 32 + lane;
        float v = row[d];
        frow[d] = (_Float16)v;
        s += v * v;
    }
#pragma unroll
    for (int off = 16; off >= 1; off >>= 1) s += __shfl_xor(s, off, 32);
    if (lane == 0) e2[k] = s;
    if (blockIdx.x == 0 && threadIdx.x == 0) *acc = 0.f;
}

// ---------------------------------------------------------------------------
// Kernel 2: one wave per tile of 16 points; block of 8 waves shares the
// codebook through LDS (staged 128 codes at a time). WMMA f32<-f16 distance
// matmul against all 1024 codes, running argmin, fused gather/STE + loss.
// ---------------------------------------------------------------------------
__global__ __launch_bounds__(256) void vq_main(const float* __restrict__ lat,
                                               const float* __restrict__ emb,
                                               const _Float16* __restrict__ ef,
                                               const float* __restrict__ e2,
                                               float* __restrict__ acc,
                                               float* __restrict__ outq,
                                               float* __restrict__ outidx) {
    __shared__ _Float16 sB[CHUNK_CODES * DIM];   // 64 KB codebook chunk (f16)
    __shared__ int bestS[8][16];

    int w    = threadIdx.x >> 5;
    int lane = threadIdx.x & 31;
    int wave = blockIdx.x * 8 + w;          // 0..4095
    int n0   = wave * 16;                   // first point of tile
    int b    = n0 >> 10;                    // batch index (HW = 1024)
    int hw0  = n0 & 1023;                   // hw offset (tile never crosses b)
    int h    = lane >> 4;                   // lane-half (K split per WMMA layout)
    int m    = lane & 15;                   // point-in-tile / code-in-tile

    const float* latbase = lat + (size_t)b * (DIM * HW) + hw0;

    // ---- Load all 8 A fragments (16 pts x 32 dims each) into registers.
    // A order used here (B matches): lane half h, element e -> K = h*16 + e
    v16h a[8];
#pragma unroll
    for (int c = 0; c < 8; ++c) {
#pragma unroll
        for (int e = 0; e < 16; ++e) {
            int d = c * 32 + h * 16 + e;
            a[c][e] = (_Float16)latbase[(size_t)d * HW + m];
        }
    }

    float bestd[8];
    int   bestc[8];
#pragma unroll
    for (int r = 0; r < 8; ++r) { bestd[r] = 3.4e38f; bestc[r] = 0; }

    // 8 chunks of 128 codes; all 8 waves share each staged chunk.
    for (int s = 0; s < NUM_EMB / CHUNK_CODES; ++s) {
        // ---- Cooperative stage: 64 KB = 4096 uint4, 16 per thread, coalesced
        __syncthreads();                     // previous chunk's compute done
        {
            const uint4* src = (const uint4*)(ef + (size_t)s * CHUNK_CODES * DIM);
            uint4*       dst = (uint4*)sB;
#pragma unroll
            for (int i = 0; i < 16; ++i)
                dst[i * 256 + threadIdx.x] = src[i * 256 + threadIdx.x];
        }
        __syncthreads();                     // chunk visible to all waves

        // ---- 8 code tiles per chunk, two in flight for WMMA ILP
        for (int tt = 0; tt < 8; tt += 2) {
            v8f acc0 = {};
            v8f acc1 = {};
            // B layout: lane = column (code), K(e) = h*16 + e -> 32 B contig
            const _Float16* b0 = sB + (size_t)(tt * 16 + m) * DIM + h * 16;
            const _Float16* b1 = b0 + (size_t)16 * DIM;
#pragma unroll
            for (int c = 0; c < 8; ++c) {
                v16h bf0 = *(const v16h*)(b0 + c * 32);
                v16h bf1 = *(const v16h*)(b1 + c * 32);
                acc0 = __builtin_amdgcn_wmma_f32_16x16x32_f16(
                    false, a[c], false, bf0, (short)0, acc0, false, false);
                acc1 = __builtin_amdgcn_wmma_f32_16x16x32_f16(
                    false, a[c], false, bf1, (short)0, acc1, false, false);
            }
            // argmin over dist = |e|^2 - 2*dot  (|x|^2 dropped: row-constant)
            int cbase = s * CHUNK_CODES + tt * 16;
            float e20 = e2[cbase + m];
            float e21 = e2[cbase + 16 + m];
#pragma unroll
            for (int r = 0; r < 8; ++r) {
                float d0 = e20 - 2.f * acc0[r];
                if (d0 < bestd[r]) { bestd[r] = d0; bestc[r] = cbase + m; }
                float d1 = e21 - 2.f * acc1[r];
                if (d1 < bestd[r]) { bestd[r] = d1; bestc[r] = cbase + 16 + m; }
            }
        }
    }

    // ---- Cross-lane min over the 16 lanes of each half (tie -> lower index)
#pragma unroll
    for (int off = 1; off <= 8; off <<= 1) {
#pragma unroll
        for (int r = 0; r < 8; ++r) {
            float od = __shfl_xor(bestd[r], off, 32);
            int   oc = __shfl_xor(bestc[r], off, 32);
            if (od < bestd[r] || (od == bestd[r] && oc < bestc[r])) {
                bestd[r] = od; bestc[r] = oc;
            }
        }
    }

    // lanes 0-15 hold winners for points 0..7 (M=r), lanes 16-31 for 8..15
    if (lane == 0) {
#pragma unroll
        for (int r = 0; r < 8; ++r) {
            bestS[w][r] = bestc[r];
            outidx[n0 + r] = (float)bestc[r];
        }
    }
    if (lane == 16) {
#pragma unroll
        for (int r = 0; r < 8; ++r) {
            bestS[w][8 + r] = bestc[r];
            outidx[n0 + 8 + r] = (float)bestc[r];
        }
    }
    __syncthreads();

    // ---- Fused gather + straight-through write + loss partial.
    // elem = iter*32 + lane -> d = iter*2 + h, point = lane&15 (coalesced)
    int code = bestS[w][m];
    const float* erow    = emb  + (size_t)code * DIM;
    float*       outbase = outq + (size_t)b * (DIM * HW) + hw0;
    float lsum = 0.f;
#pragma unroll 4
    for (int iter = 0; iter < 128; ++iter) {
        int d   = iter * 2 + h;
        float q = erow[d];
        float l = latbase[(size_t)d * HW + m];
        float df = q - l;
        lsum += df * df;
        outbase[(size_t)d * HW + m] = l + (q - l);   // STE, mimics reference
    }
#pragma unroll
    for (int off = 16; off >= 1; off >>= 1) lsum += __shfl_xor(lsum, off, 32);
    if (lane == 0) atomicAdd(acc, lsum);
}

// ---------------------------------------------------------------------------
// Kernel 3: loss = (1 + 0.25) * sum / numel
// ---------------------------------------------------------------------------
__global__ void vq_fin(const float* __restrict__ acc, float* __restrict__ outloss) {
    *outloss = 1.25f * (*acc) / 16777216.f;
}

extern "C" void kernel_launch(void* const* d_in, const int* in_sizes, int n_in,
                              void* d_out, int out_size, void* d_ws, size_t ws_size,
                              hipStream_t stream) {
    const float* lat = (const float*)d_in[0];   // [64,256,32,32]
    const float* emb = (const float*)d_in[1];   // [1024,256]

    float* outq    = (float*)d_out;             // quant_st  (16777216)
    float* outloss = outq + (size_t)QELEMS;     // loss      (1)
    float* outidx  = outloss + 1;               // indices   (65536, as f32)

    char*     ws  = (char*)d_ws;
    float*    acc = (float*)ws;                          // 4 B
    float*    e2  = (float*)(ws + 256);                  // 4 KB
    _Float16* ef  = (_Float16*)(ws + 256 + 4096);        // 512 KB packed f16

    vq_prep<<<128, 256, 0, stream>>>(emb, e2, ef, acc);
    vq_main<<<512, 256, 0, stream>>>(lat, emb, ef, e2, acc, outq, outidx);
    vq_fin<<<1, 1, 0, stream>>>(acc, outloss);
}
